// MultiHeadSelfAttention_59261958750314
// MI455X (gfx1250) — matrix operand order repaired and checked by reading, not compiled, not run on hardware
//
#include <hip/hip_runtime.h>
#include <hip/hip_bf16.h>

typedef _Float16 half_t;
typedef __attribute__((ext_vector_type(16))) _Float16 v16h;
typedef __attribute__((ext_vector_type(8)))  _Float16 v8h;
typedef __attribute__((ext_vector_type(8)))  float    v8f;
typedef __attribute__((ext_vector_type(4)))  int      v4i;

#define B_  4
#define L_  2048
#define E_  1024
#define H_  16
#define D_  64
#define N3_ 3072
#define M_  (B_*L_)   // 8192 rows of x

// LDS staging geometry for the GEMMs
#define A_ROWS   128            // block rows (M)
#define B_COLS   256            // block cols (N)
#define SA       40             // padded LDS row stride in halfs (80 B) -> conflict-free
#define AHALF    (A_ROWS * SA)  // 5120 halfs
#define BHALF    (B_COLS * SA)  // 10240 halfs
#define BUFHALF  (AHALF + BHALF)

union AFrag { v16h v; v8h h[2]; };

static __device__ __forceinline__ v8f wmma_f16(const AFrag& a, const AFrag& b, v8f c) {
    return __builtin_amdgcn_wmma_f32_16x16x32_f16(false, a.v, false, b.v, (short)0, c, false, false);
}

static __device__ __forceinline__ v8f vzero8() {
    v8f z = {0.f, 0.f, 0.f, 0.f, 0.f, 0.f, 0.f, 0.f};
    return z;
}

// A-fragment (16x32 f16): lane<16 holds K base+0..7 and base+16..23
static __device__ __forceinline__ void load_afrag(const half_t* p, AFrag& f) {
    f.h[0] = *(const v8h*)(p);
    f.h[1] = *(const v8h*)(p + 16);
}
// B-fragment (32x16 f16): lane<16 holds K base+0..7 and base+16..23 of column lane&15
static __device__ __forceinline__ void load_bfrag(const half_t* p, AFrag& f) {
    f.h[0] = *(const v8h*)(p);
    f.h[1] = *(const v8h*)(p + 16);
}

// ---- async global -> LDS (ASYNCcnt path), with sync fallback ----
static __device__ __forceinline__ void async_b128(const half_t* g, half_t* l) {
#if __has_builtin(__builtin_amdgcn_global_load_async_to_lds_b128)
    __builtin_amdgcn_global_load_async_to_lds_b128(
        (__attribute__((address_space(1))) v4i*)(g),
        (__attribute__((address_space(3))) v4i*)(l), 0, 0);
#else
    *(v8h*)l = *(const v8h*)g;
#endif
}

static __device__ __forceinline__ void wait_async0() {
#if __has_builtin(__builtin_amdgcn_s_wait_asynccnt)
    __builtin_amdgcn_s_wait_asynccnt(0);
#else
    asm volatile("s_wait_asynccnt 0x0" ::: "memory");
#endif
}

// ---------------- conversion / layout kernels ----------------

__global__ void cvt_f32_f16(const float* __restrict__ in, half_t* __restrict__ out, int n) {
    int i = blockIdx.x * blockDim.x + threadIdx.x;
    int stride = gridDim.x * blockDim.x;
    for (; i < n; i += stride) out[i] = (half_t)in[i];
}

// in: [rows, cols] f32 row-major  ->  out: [cols, rows] f16 row-major (i.e. B^T)
__global__ void transpose_cvt(const float* __restrict__ in, half_t* __restrict__ out,
                              int rows, int cols) {
    int i = blockIdx.x * blockDim.x + threadIdx.x;
    int n = rows * cols;
    int stride = gridDim.x * blockDim.x;
    for (; i < n; i += stride) {
        int c = i / rows;
        int r = i - c * rows;
        out[i] = (half_t)in[(size_t)r * cols + c];
    }
}

// -------- LDS-staged double-buffered GEMM core: block 128x256, wave 64x64 --------
// A: [M,K] f16 row-major.  Bt: [N,K] f16 row-major (column-major B).
__device__ __forceinline__ void gemm_staged(const half_t* __restrict__ A,
                                            const half_t* __restrict__ Bt,
                                            int K, int rowBase0, int colBase0,
                                            v8f (&acc)[4][4]) {
    __shared__ __align__(16) half_t lds[2 * BUFHALF];   // 60 KB: A/B tiles, 2 buffers

    const int tid  = threadIdx.x;
    const int lane = tid & 31;
    const int wave = tid >> 5;
    const int waveM = wave >> 2, waveN = wave & 3;       // 2 x 4 waves
    const int rsel = lane >> 4, lidx = lane & 15;

    // cooperative tile fill: per thread 2 async b128 for A, 4 for B
    auto fill = [&](int buf, int k) {
        half_t* ab = lds + buf * BUFHALF;
        half_t* bb = ab + AHALF;
        {
            int r = tid >> 1, seg = tid & 1;             // 128 rows x 2 halves
            const half_t* g = A + (size_t)(rowBase0 + r) * K + k + seg * 16;
            half_t* l = ab + r * SA + seg * 16;
            async_b128(g, l);
            async_b128(g + 8, l + 8);
        }
        {
            const half_t* g = Bt + (size_t)(colBase0 + tid) * K + k;   // 256 cols
            half_t* l = bb + tid * SA;
            async_b128(g,      l);
            async_b128(g + 8,  l + 8);
            async_b128(g + 16, l + 16);
            async_b128(g + 24, l + 24);
        }
    };

    fill(0, 0);
    wait_async0();
    __syncthreads();

    for (int k = 0; k < K; k += 32) {
        const int buf = (k >> 5) & 1;
        if (k + 32 < K) fill(buf ^ 1, k + 32);
        if (k + 128 < K) {   // deep prefetch into L2 (global_prefetch_b8)
            __builtin_prefetch(A + (size_t)(rowBase0 + (tid >> 1)) * K + k + 128, 0, 1);
            __builtin_prefetch(Bt + (size_t)(colBase0 + tid) * K + k + 128, 0, 1);
        }

        const half_t* ab = lds + buf * BUFHALF;
        const half_t* bb = ab + AHALF;

        AFrag a[4], b[4];
#pragma unroll
        for (int mi = 0; mi < 4; ++mi) {
            const half_t* p = ab + (waveM * 64 + mi * 16 + lidx) * SA + rsel * 8;
            a[mi].h[0] = *(const v8h*)(p);
            a[mi].h[1] = *(const v8h*)(p + 16);
        }
#pragma unroll
        for (int ni = 0; ni < 4; ++ni) {
            const half_t* p = bb + (waveN * 64 + ni * 16 + lidx) * SA + rsel * 8;
            b[ni].h[0] = *(const v8h*)(p);
            b[ni].h[1] = *(const v8h*)(p + 16);
        }
#pragma unroll
        for (int mi = 0; mi < 4; ++mi)
#pragma unroll
            for (int ni = 0; ni < 4; ++ni)
                acc[mi][ni] = wmma_f16(a[mi], b[ni], acc[mi][ni]);

        wait_async0();      // next buffer's DMA complete (overlapped with the WMMAs)
        __syncthreads();    // all waves done reading current buffer
    }
}

// ---------------- QKV GEMM: [8192,1024] x [1024,3072] ----------------
__global__ __launch_bounds__(256)
void gemm_qkv(const half_t* __restrict__ xh, const half_t* __restrict__ wT,
              half_t* __restrict__ qout, half_t* __restrict__ kout,
              half_t* __restrict__ vtout) {
    const int lane = threadIdx.x & 31;
    const int wave = threadIdx.x >> 5;
    const int waveM = wave >> 2, waveN = wave & 3;
    const int rowBase0 = blockIdx.y * A_ROWS;
    const int colBase0 = blockIdx.x * B_COLS;

    v8f acc[4][4];
#pragma unroll
    for (int mi = 0; mi < 4; ++mi)
#pragma unroll
        for (int ni = 0; ni < 4; ++ni) acc[mi][ni] = vzero8();

    gemm_staged(xh, wT, E_, rowBase0, colBase0, acc);

    const int rsel = lane >> 4, lidx = lane & 15;
#pragma unroll
    for (int mi = 0; mi < 4; ++mi)
#pragma unroll
        for (int ni = 0; ni < 4; ++ni)
#pragma unroll
            for (int i = 0; i < 8; ++i) {
                int m = rowBase0 + waveM * 64 + mi * 16 + rsel * 8 + i;
                int n = colBase0 + waveN * 64 + ni * 16 + lidx;
                int b  = m >> 11;            // /L_
                int l  = m & (L_ - 1);
                int wh = n >> 10;            // 0:Q 1:K 2:V
                int r  = n & (E_ - 1);
                int h  = r >> 6;             // /D_
                int d  = r & (D_ - 1);
                half_t val = (half_t)acc[mi][ni][i];
                size_t bh = (size_t)b * H_ + h;
                if (wh == 0)      qout [(bh * L_ + l) * D_ + d] = val;
                else if (wh == 1) kout [(bh * L_ + l) * D_ + d] = val;
                else              vtout[(bh * D_ + d) * L_ + l] = val;
            }
}

// ---------------- flash attention: one wave per 16 queries, pipelined keys ------

__device__ __forceinline__ void attn_load_k(const half_t* __restrict__ kp,
                                            int j, int rsel, int lidx,
                                            AFrag (&kb)[2][2]) {
#pragma unroll
    for (int ds = 0; ds < 2; ++ds)
#pragma unroll
        for (int ni = 0; ni < 2; ++ni)
            load_bfrag(kp + (size_t)(j + ni * 16 + lidx) * D_ + ds * 32 + rsel * 8,
                       kb[ds][ni]);
}

__device__ __forceinline__ void attn_step(const half_t* __restrict__ kp,
                                          const half_t* __restrict__ vp,
                                          half_t* __restrict__ lp,
                                          const AFrag (&qa)[2],
                                          AFrag (&kcur)[2][2], AFrag (&knxt)[2][2],
                                          int j, bool pref, int rsel, int lidx,
                                          v8f (&o)[4], float (&rmax)[8], float (&rsum)[8]) {
    if (pref) attn_load_k(kp, j + 32, rsel, lidx, knxt);

    AFrag vb[4];
#pragma unroll
    for (int t = 0; t < 4; ++t)
        load_bfrag(vp + (size_t)(t * 16 + lidx) * L_ + j + rsel * 8, vb[t]);

    v8f s[2];
    s[0] = vzero8(); s[1] = vzero8();
#pragma unroll
    for (int ds = 0; ds < 2; ++ds)
#pragma unroll
        for (int ni = 0; ni < 2; ++ni)
            s[ni] = wmma_f16(qa[ds], kcur[ds][ni], s[ni]);

    float p0[8], p1[8];
#pragma unroll
    for (int i = 0; i < 8; ++i) {
        float s0 = s[0][i] * 0.03125f;
        float s1 = s[1][i] * 0.03125f;
        float tm = fmaxf(s0, s1);
        tm = fmaxf(tm, __shfl_xor(tm, 1, 32));
        tm = fmaxf(tm, __shfl_xor(tm, 2, 32));
        tm = fmaxf(tm, __shfl_xor(tm, 4, 32));
        tm = fmaxf(tm, __shfl_xor(tm, 8, 32));
        float nm = fmaxf(rmax[i], tm);
        float sc = __expf(rmax[i] - nm);
        rmax[i] = nm;
        float e0 = __expf(s0 - nm);
        float e1 = __expf(s1 - nm);
        p0[i] = e0; p1[i] = e1;
        float ts = e0 + e1;
        ts += __shfl_xor(ts, 1, 32);
        ts += __shfl_xor(ts, 2, 32);
        ts += __shfl_xor(ts, 4, 32);
        ts += __shfl_xor(ts, 8, 32);
        rsum[i] = rsum[i] * sc + ts;
#pragma unroll
        for (int t = 0; t < 4; ++t) o[t][i] *= sc;
    }

#pragma unroll
    for (int i = 0; i < 8; ++i) {
        int m = rsel * 8 + i;
        lp[m * 32 + lidx]      = (half_t)p0[i];
        lp[m * 32 + 16 + lidx] = (half_t)p1[i];
    }
    asm volatile("s_wait_dscnt 0x0" ::: "memory");
    AFrag pa;
    load_afrag(lp + lidx * 32 + rsel * 8, pa);

#pragma unroll
    for (int t = 0; t < 4; ++t)
        o[t] = wmma_f16(pa, vb[t], o[t]);
    asm volatile("s_wait_dscnt 0x0" ::: "memory");
}

__global__ __launch_bounds__(128)
void attn(const half_t* __restrict__ q, const half_t* __restrict__ kmat,
          const half_t* __restrict__ vt, half_t* __restrict__ attout) {
    __shared__ half_t plds[4][16 * 32];
    const int lane = threadIdx.x & 31;
    const int wave = threadIdx.x >> 5;
    const int bh = blockIdx.y;
    const int b = bh >> 4, h = bh & (H_ - 1);
    const int qBase = blockIdx.x * 64 + wave * 16;

    const half_t* qp = q    + (size_t)bh * L_ * D_;
    const half_t* kp = kmat + (size_t)bh * L_ * D_;
    const half_t* vp = vt   + (size_t)bh * D_ * L_;
    half_t* lp = plds[wave];

    const int rsel = lane >> 4, lidx = lane & 15;

    AFrag qa[2];
#pragma unroll
    for (int s = 0; s < 2; ++s)
        load_afrag(qp + (size_t)(qBase + lidx) * D_ + s * 32 + rsel * 8, qa[s]);

    v8f o[4];
#pragma unroll
    for (int t = 0; t < 4; ++t) o[t] = vzero8();
    float rmax[8], rsum[8];
#pragma unroll
    for (int i = 0; i < 8; ++i) { rmax[i] = -3.0e38f; rsum[i] = 0.f; }

    AFrag kb0[2][2], kb1[2][2];
    attn_load_k(kp, 0, rsel, lidx, kb0);
    for (int j = 0; j < L_; j += 64) {
        attn_step(kp, vp, lp, qa, kb0, kb1, j,      true,          rsel, lidx, o, rmax, rsum);
        attn_step(kp, vp, lp, qa, kb1, kb0, j + 32, (j + 64) < L_, rsel, lidx, o, rmax, rsum);
    }

#pragma unroll
    for (int t = 0; t < 4; ++t)
#pragma unroll
        for (int i = 0; i < 8; ++i) {
            int qg = qBase + rsel * 8 + i;
            int d  = t * 16 + lidx;
            float val = o[t][i] / rsum[i];
            attout[((size_t)b * L_ + qg) * E_ + h * D_ + d] = (half_t)val;
        }
}

// ---------------- output projection GEMM + bias -> f32 d_out ----------------
__global__ __launch_bounds__(256)
void gemm_proj(const half_t* __restrict__ att, const half_t* __restrict__ wT,
               const float* __restrict__ bias, float* __restrict__ out) {
    const int lane = threadIdx.x & 31;
    const int wave = threadIdx.x >> 5;
    const int waveM = wave >> 2, waveN = wave & 3;
    const int rowBase0 = blockIdx.y * A_ROWS;
    const int colBase0 = blockIdx.x * B_COLS;

    v8f acc[4][4];
#pragma unroll
    for (int mi = 0; mi < 4; ++mi)
#pragma unroll
        for (int ni = 0; ni < 4; ++ni) acc[mi][ni] = vzero8();

    gemm_staged(att, wT, E_, rowBase0, colBase0, acc);

    const int rsel = lane >> 4, lidx = lane & 15;
#pragma unroll
    for (int mi = 0; mi < 4; ++mi)
#pragma unroll
        for (int ni = 0; ni < 4; ++ni) {
            int n = colBase0 + waveN * 64 + ni * 16 + lidx;
            float bv = bias[n];
#pragma unroll
            for (int i = 0; i < 8; ++i) {
                int m = rowBase0 + waveM * 64 + mi * 16 + rsel * 8 + i;
                out[(size_t)m * E_ + n] = acc[mi][ni][i] + bv;
            }
        }
}

// ---------------- host launch ----------------
extern "C" void kernel_launch(void* const* d_in, const int* in_sizes, int n_in,
                              void* d_out, int out_size, void* d_ws, size_t ws_size,
                              hipStream_t stream) {
    const float* x      = (const float*)d_in[0];
    const float* w_qkv  = (const float*)d_in[1];
    const float* w_proj = (const float*)d_in[2];
    const float* b_proj = (const float*)d_in[3];
    float* out = (float*)d_out;

    char* ws = (char*)d_ws;
    half_t* xh  = (half_t*)ws; ws += (size_t)M_ * E_ * sizeof(half_t);
    half_t* wqT = (half_t*)ws; ws += (size_t)N3_ * E_ * sizeof(half_t);
    half_t* wpT = (half_t*)ws; ws += (size_t)E_ * E_ * sizeof(half_t);
    half_t* qh  = (half_t*)ws; ws += (size_t)M_ * E_ * sizeof(half_t);
    half_t* kh  = (half_t*)ws; ws += (size_t)M_ * E_ * sizeof(half_t);
    half_t* vth = (half_t*)ws; ws += (size_t)M_ * E_ * sizeof(half_t);
    half_t* ath = (half_t*)ws; ws += (size_t)M_ * E_ * sizeof(half_t);

    cvt_f32_f16<<<2048, 256, 0, stream>>>(x, xh, M_ * E_);
    transpose_cvt<<<2048, 256, 0, stream>>>(w_qkv, wqT, E_, N3_);
    transpose_cvt<<<1024, 256, 0, stream>>>(w_proj, wpT, E_, E_);

    gemm_qkv<<<dim3(N3_ / B_COLS, M_ / A_ROWS), 256, 0, stream>>>(xh, wqT, qh, kh, vth);
    attn<<<dim3(L_ / 64, B_ * H_), 128, 0, stream>>>(qh, kh, vth, ath);
    gemm_proj<<<dim3(E_ / B_COLS, M_ / A_ROWS), 256, 0, stream>>>(ath, wpT, b_proj, out);
}
